// MultiheadAttention_39359080301188
// MI455X (gfx1250) — matrix operand order repaired and checked by reading, not compiled, not run on hardware
//
#include <hip/hip_runtime.h>
#include <hip/hip_bf16.h>
#include <math.h>

// Problem constants (from the reference)
#define B_   2
#define C_   512
#define T_   4096
#define H_   8
#define D_   64
#define OD_  512   // HEADS * DIM_HEAD

typedef __bf16 bf16;
typedef __attribute__((ext_vector_type(16))) __bf16 v16bf;
typedef __attribute__((ext_vector_type(8)))  __bf16 v8bf;
typedef __attribute__((ext_vector_type(8)))  float  v8f;

static __device__ __forceinline__ v8f wmma_bf16(v16bf a, v16bf b, v8f c) {
  // D = A(16x32 bf16) * B(32x16 bf16) + C(16x16 f32)
  return __builtin_amdgcn_wmma_f32_16x16x32_bf16(false, a, false, b, (short)0, c,
                                                 false, false);
}

// A fragment (16x32, bf16, row-major source, leading dim `ld` in elements).
// CDNA5 layout: lane row = lane&15; lanes 0-15 hold K=0..7 (v0-3) and
// K=16..23 (v4-7); lanes 16-31 hold K=8..15 and K=24..31.
static __device__ __forceinline__ v16bf load_a_frag(const bf16* base, int ld, int lane) {
  const int row = lane & 15;
  const int kb  = (lane >> 4) * 8;
  const bf16* p = base + (size_t)row * ld + kb;
  v8bf lo = *(const v8bf*)(p);
  v8bf hi = *(const v8bf*)(p + 16);
  v16bf r;
#pragma unroll
  for (int i = 0; i < 8; ++i) { r[i] = lo[i]; r[i + 8] = hi[i]; }
  return r;
}

// B fragment (32x16): B[k][n] = R[n][k] with R row-major, 16 "n" rows,
// 32 contiguous k values.  lane col n = lane&15; lanes 0-15 hold K=0..7 and
// K=16..23; lanes 16-31 hold K=8..15 and K=24..31 (same as A).
static __device__ __forceinline__ v16bf load_b_frag(const bf16* base, int ld, int lane) {
  const int n  = lane & 15;
  const int kb = (lane >> 4) * 8;
  const bf16* p = base + (size_t)n * ld + kb;
  v8bf lo = *(const v8bf*)(p);
  v8bf hi = *(const v8bf*)(p + 16);
  v16bf r;
#pragma unroll
  for (int i = 0; i < 8; ++i) { r[i] = lo[i]; r[i + 8] = hi[i]; }
  return r;
}

// Async 16B copy global -> LDS (GVS mode: SGPR base + signed 32-bit VGPR
// offset).  Tracked by ASYNCcnt.  See CDNA5 ISA.
static __device__ __forceinline__ void async_copy_b128(uint32_t lds_addr,
                                                       uint64_t sbase,
                                                       uint32_t voff) {
  asm volatile("global_load_async_to_lds_b128 %0, %1, %2"
               :: "v"(lds_addr), "v"(voff), "s"(sbase) : "memory");
}
static __device__ __forceinline__ void wait_async0() {
  asm volatile("s_wait_asynccnt 0x0" ::: "memory");
}

// ---------------------------------------------------------------------------
// Pre-pass A: X (B,C,T) f32  ->  Xb (B*T, C) bf16, LDS-tiled 32x32 transpose,
// coalesced on both the read (along t) and the write (along c).
// ---------------------------------------------------------------------------
__global__ __launch_bounds__(256) void transpose_x_kernel(
    const float* __restrict__ X, bf16* __restrict__ Xb) {
  __shared__ float tile[32][33];
  const int b   = blockIdx.z;
  const int c0  = blockIdx.y * 32;
  const int t0  = blockIdx.x * 32;
  const int col = threadIdx.x & 31;
  const int r8  = threadIdx.x >> 5;       // 0..7
#pragma unroll
  for (int i = 0; i < 4; ++i) {
    const int r = r8 + i * 8;             // c offset
    tile[r][col] = X[((size_t)b * C_ + (c0 + r)) * T_ + (t0 + col)];
  }
  __syncthreads();
#pragma unroll
  for (int i = 0; i < 4; ++i) {
    const int r = r8 + i * 8;             // t offset
    Xb[((size_t)b * T_ + (t0 + r)) * C_ + (c0 + col)] = (bf16)tile[col][r];
  }
}

// ---------------------------------------------------------------------------
// Pre-pass B: weights (K,N) f32 -> (N,K) bf16 (B-fragment layout).
// grid.z selects Wq / Wk / Wv / Wo; all are 512x512.
// ---------------------------------------------------------------------------
__global__ __launch_bounds__(256) void transpose_w_kernel(
    const float* __restrict__ Wq, const float* __restrict__ Wk,
    const float* __restrict__ Wv, const float* __restrict__ Wo,
    bf16* __restrict__ Wt) {    // 4 buffers of 512*512, concatenated
  __shared__ float tile[32][33];
  const int wsel = blockIdx.z;
  const float* W = (wsel == 0) ? Wq : (wsel == 1) ? Wk : (wsel == 2) ? Wv : Wo;
  bf16* dst = Wt + (size_t)wsel * OD_ * OD_;

  const int k0  = blockIdx.y * 32;        // source row (k)
  const int n0  = blockIdx.x * 32;        // source col (n)
  const int col = threadIdx.x & 31;
  const int r8  = threadIdx.x >> 5;
#pragma unroll
  for (int i = 0; i < 4; ++i) {
    const int r = r8 + i * 8;             // k offset
    tile[r][col] = W[(size_t)(k0 + r) * OD_ + (n0 + col)];
  }
  __syncthreads();
#pragma unroll
  for (int i = 0; i < 4; ++i) {
    const int r = r8 + i * 8;             // n offset
    dst[(size_t)(n0 + r) * OD_ + (k0 + col)] = (bf16)tile[col][r];
  }
}

// ---------------------------------------------------------------------------
// Kernel 1: QKV projections.  Y = Xb @ W, all-bf16 vector-load operands.
// One wave -> 32 (t) x 64 (one head's d) tile: 2 A frags x 4 B frags
// -> 8 WMMAs per 32-k step.  grid.z selects Q/K/V.
// Q,K stored (b,h,t,d) bf16; V stored transposed (b,h,d,t) bf16.
// ---------------------------------------------------------------------------
__global__ __launch_bounds__(128) void qkv_kernel(
    const bf16* __restrict__ Xb, const bf16* __restrict__ Wt,
    bf16* __restrict__ Q, bf16* __restrict__ K, bf16* __restrict__ Vt) {
  const int lane = threadIdx.x & 31;
  const int wv   = threadIdx.x >> 5;
  const int mt   = blockIdx.x * 4 + wv;   // 32-row tile index in M = B*T
  const int head = blockIdx.y;            // 64-col tile == one head
  const int wsel = blockIdx.z;            // 0=Q 1=K 2=V
  const bf16* W  = Wt + (size_t)wsel * OD_ * OD_;   // (N,K) bf16

  const int m0 = mt * 32;
  const int b  = m0 / T_;
  const int t0 = m0 % T_;                 // multiple of 32: no batch straddle
  const int n0 = head * D_;

  const bf16* Arow0 = Xb + (size_t)m0 * C_;
  const bf16* Arow1 = Arow0 + (size_t)16 * C_;

  v8f acc[2][4] = {};
  for (int cb = 0; cb < C_; cb += 32) {
    const v16bf a0 = load_a_frag(Arow0 + cb, C_, lane);
    const v16bf a1 = load_a_frag(Arow1 + cb, C_, lane);
#pragma unroll
    for (int nt = 0; nt < 4; ++nt) {
      const v16bf bb = load_b_frag(W + (size_t)(n0 + nt * 16) * OD_ + cb, OD_, lane);
      acc[0][nt] = wmma_bf16(a0, bb, acc[0][nt]);
      acc[1][nt] = wmma_bf16(a1, bb, acc[1][nt]);
    }
  }

  // C/D layout: row = (lane>>4)*8 + r, col = nt*16 + (lane&15)
  const int rowb = (lane >> 4) * 8;
  const int col  = lane & 15;
#pragma unroll
  for (int half = 0; half < 2; ++half) {
#pragma unroll
    for (int nt = 0; nt < 4; ++nt) {
#pragma unroll
      for (int r = 0; r < 8; ++r) {
        const int t = t0 + half * 16 + rowb + r;
        const int d = nt * 16 + col;
        const float v = acc[half][nt][r];
        if (wsel == 0)
          Q [(((size_t)b * H_ + head) * T_ + t) * D_ + d] = (bf16)v;
        else if (wsel == 1)
          K [(((size_t)b * H_ + head) * T_ + t) * D_ + d] = (bf16)v;
        else
          Vt[(((size_t)b * H_ + head) * D_ + d) * T_ + t] = (bf16)v;
      }
    }
  }
}

// ---------------------------------------------------------------------------
// Kernel 2: flash attention with async-LDS staging of K / V^T.
// Block = 8 waves = 128 queries of one (b,h).  Each 32-key block of K
// (32x64 bf16, 4KB) and V^T (64x32 bf16, 4KB) is copied to LDS ONCE per
// workgroup with GLOBAL_LOAD_ASYNC_TO_LDS_B128 (double-buffered, ASYNCcnt),
// instead of each of the 8 waves loading it redundantly from L2.
// Per wave: QK^T = 4 WMMAs, online softmax (shfl_xor within 16-lane halves,
// matching the C/D row split), P transposed through LDS to A layout,
// PV = 4 WMMAs.
// ---------------------------------------------------------------------------
__global__ __launch_bounds__(256) void attn_kernel(
    const bf16* __restrict__ Q, const bf16* __restrict__ K,
    const bf16* __restrict__ Vt, bf16* __restrict__ O) {
  __shared__ bf16 ldsK[2][32 * 64];       // 2 x 4KB : rows = keys, d contig
  __shared__ bf16 ldsV[2][64 * 32];       // 2 x 4KB : rows = d, keys contig
  __shared__ bf16 ldsP[8][16 * 32];       // 8KB per-wave P staging

  const int tid  = threadIdx.x;
  const int lane = tid & 31;
  const int wv   = tid >> 5;
  const int bh   = blockIdx.y;            // b*H + h
  const int q0   = blockIdx.x * 128 + wv * 16;

  const bf16* Qb = Q  + ((size_t)bh * T_ + q0) * D_;
  const uint64_t Kbase = (uint64_t)(uintptr_t)(K  + (size_t)bh * T_ * D_);
  const uint64_t Vbase = (uint64_t)(uintptr_t)(Vt + (size_t)bh * D_ * T_);

  // per-thread async-copy slots: 16B of K and 16B of V^T per 32-key block
  const int krow = tid >> 3, kseg = tid & 7;   // K: 32 rows x 128B
  const int vrow = tid >> 2, vseg = tid & 3;   // V: 64 rows x 64B
  const uint32_t ldsKaddr[2] = {
      (uint32_t)(uintptr_t)&ldsK[0][krow * 64 + kseg * 8],
      (uint32_t)(uintptr_t)&ldsK[1][krow * 64 + kseg * 8]};
  const uint32_t ldsVaddr[2] = {
      (uint32_t)(uintptr_t)&ldsV[0][vrow * 32 + vseg * 8],
      (uint32_t)(uintptr_t)&ldsV[1][vrow * 32 + vseg * 8]};

  // stage key-block starting at kb into buffer buf
#define STAGE_KV(buf, kb)                                                      \
  do {                                                                         \
    async_copy_b128(ldsKaddr[buf], Kbase,                                      \
                    (uint32_t)((((kb) + krow) * D_ + kseg * 8) * 2));          \
    async_copy_b128(ldsVaddr[buf], Vbase,                                      \
                    (uint32_t)((vrow * T_ + (kb) + vseg * 8) * 2));            \
  } while (0)

  // Q fragments live in registers for the whole key loop (d = 0..31, 32..63)
  const v16bf aq0 = load_a_frag(Qb + 0,  D_, lane);
  const v16bf aq1 = load_a_frag(Qb + 32, D_, lane);

  v8f o[4] = {};
  float m_i[8], l_i[8];
#pragma unroll
  for (int r = 0; r < 8; ++r) { m_i[r] = -INFINITY; l_i[r] = 0.0f; }

  const int prow = (lane >> 4) * 8;       // C/D row base held by this lane
  const int pcol = lane & 15;

  STAGE_KV(0, 0);
  for (int kb = 0; kb < T_; kb += 32) {
    const int buf = (kb >> 5) & 1;
    wait_async0();                        // this wave's copies landed
    __syncthreads();                      // all waves' copies landed
    if (kb + 32 < T_) STAGE_KV(buf ^ 1, kb + 32);

    const bf16* Kl = &ldsK[buf][0];
    const bf16* Vl = &ldsV[buf][0];

    // scores: two 16x16 tiles over this 32-key block (B frags from LDS)
    v8f s0 = {}, s1 = {};
    {
      v16bf bk;
      bk = load_b_frag(Kl +  0 * 64 + 0,  64, lane);
      s0 = wmma_bf16(aq0, bk, s0);
      bk = load_b_frag(Kl +  0 * 64 + 32, 64, lane);
      s0 = wmma_bf16(aq1, bk, s0);
      bk = load_b_frag(Kl + 16 * 64 + 0,  64, lane);
      s1 = wmma_bf16(aq0, bk, s1);
      bk = load_b_frag(Kl + 16 * 64 + 32, 64, lane);
      s1 = wmma_bf16(aq1, bk, s1);
    }

#pragma unroll
    for (int r = 0; r < 8; ++r) {
      const float x0 = s0[r] * 0.125f;    // 1/sqrt(64)
      const float x1 = s1[r] * 0.125f;
      float mb = fmaxf(x0, x1);
#pragma unroll
      for (int off = 1; off < 16; off <<= 1)
        mb = fmaxf(mb, __shfl_xor(mb, off, 32));
      const float mnew = fmaxf(m_i[r], mb);
      const float cf   = __expf(m_i[r] - mnew);
      const float p0   = __expf(x0 - mnew);
      const float p1   = __expf(x1 - mnew);
      float ls = p0 + p1;
#pragma unroll
      for (int off = 1; off < 16; off <<= 1)
        ls += __shfl_xor(ls, off, 32);
      m_i[r] = mnew;
      l_i[r] = l_i[r] * cf + ls;
#pragma unroll
      for (int nt = 0; nt < 4; ++nt) o[nt][r] *= cf;

      // C-layout -> row-major P tile in this wave's LDS region
      ldsP[wv][(prow + r) * 32 + pcol]      = (bf16)p0;
      ldsP[wv][(prow + r) * 32 + 16 + pcol] = (bf16)p1;
    }

    // P (16x32) as A fragment, V^T rows as B fragments; 4 output d-tiles
    const v16bf ap = load_a_frag(&ldsP[wv][0], 32, lane);
#pragma unroll
    for (int nt = 0; nt < 4; ++nt) {
      v16bf bv = load_b_frag(Vl + (nt * 16) * 32, 32, lane);
      o[nt] = wmma_bf16(ap, bv, o[nt]);
    }
  }
#undef STAGE_KV

  // normalize and write attention output row-major (b, t, h*64+d), bf16
  const int b = bh / H_;
  const int h = bh % H_;
#pragma unroll
  for (int nt = 0; nt < 4; ++nt) {
#pragma unroll
    for (int r = 0; r < 8; ++r) {
      const int row = prow + r;
      const float v = o[nt][r] / l_i[r];
      O[((size_t)b * T_ + q0 + row) * OD_ + (h * D_ + nt * 16 + pcol)] = (bf16)v;
    }
  }
}

// ---------------------------------------------------------------------------
// Kernel 3: output projection  out = attn @ Wo + bo, stored transposed
// into (B, 512, T) fp32.  One wave = 32 (t) x 64 (cols) tile, bf16 Wo^T.
// ---------------------------------------------------------------------------
__global__ __launch_bounds__(128) void proj_kernel(
    const bf16* __restrict__ Ain, const bf16* __restrict__ Wot,
    const float* __restrict__ bo, float* __restrict__ Out) {
  const int lane = threadIdx.x & 31;
  const int wv   = threadIdx.x >> 5;
  const int mt   = blockIdx.x * 4 + wv;
  const int n0   = blockIdx.y * 64;
  const int m0   = mt * 32;
  const int b    = m0 / T_;
  const int t0   = m0 % T_;

  const bf16* Arow0 = Ain + (size_t)m0 * OD_;
  const bf16* Arow1 = Arow0 + (size_t)16 * OD_;

  v8f acc[2][4] = {};
  for (int cb = 0; cb < OD_; cb += 32) {
    const v16bf a0 = load_a_frag(Arow0 + cb, OD_, lane);
    const v16bf a1 = load_a_frag(Arow1 + cb, OD_, lane);
#pragma unroll
    for (int nt = 0; nt < 4; ++nt) {
      const v16bf bb = load_b_frag(Wot + (size_t)(n0 + nt * 16) * OD_ + cb, OD_, lane);
      acc[0][nt] = wmma_bf16(a0, bb, acc[0][nt]);
      acc[1][nt] = wmma_bf16(a1, bb, acc[1][nt]);
    }
  }

  const int rowb = (lane >> 4) * 8;
  const int bn   = lane & 15;
#pragma unroll
  for (int half = 0; half < 2; ++half) {
#pragma unroll
    for (int nt = 0; nt < 4; ++nt) {
      const int col  = n0 + nt * 16 + bn;
      const float bias = bo[col];
#pragma unroll
      for (int r = 0; r < 8; ++r) {
        const int t = t0 + half * 16 + rowb + r;
        Out[((size_t)b * OD_ + col) * T_ + t] = acc[half][nt][r] + bias;
      }
    }
  }
}

// ---------------------------------------------------------------------------
extern "C" void kernel_launch(void* const* d_in, const int* in_sizes, int n_in,
                              void* d_out, int out_size, void* d_ws, size_t ws_size,
                              hipStream_t stream) {
  const float* X  = (const float*)d_in[0];
  const float* Wq = (const float*)d_in[1];
  const float* Wk = (const float*)d_in[2];
  const float* Wv = (const float*)d_in[3];
  const float* Wo = (const float*)d_in[4];
  const float* bo = (const float*)d_in[5];
  float* out = (float*)d_out;

  // workspace layout (bf16):
  //   Q, K (b,h,t,d); V^T (b,h,d,t); attn (b,t,512); Xb (b*t, c); Wt[4] (n,k)
  const size_t nPerBuf = (size_t)B_ * H_ * T_ * D_;   // 4,194,304 elems
  bf16* Q    = (bf16*)d_ws;
  bf16* K    = Q    + nPerBuf;
  bf16* Vt   = K    + nPerBuf;
  bf16* attn = Vt   + nPerBuf;
  bf16* Xb   = attn + nPerBuf;
  bf16* Wt   = Xb   + (size_t)B_ * T_ * C_;           // 4 x 512*512
  bf16* Wot  = Wt   + (size_t)3 * OD_ * OD_;

  {
    dim3 grid(T_ / 32, C_ / 32, B_);        // (128, 16, 2)
    transpose_x_kernel<<<grid, 256, 0, stream>>>(X, Xb);
  }
  {
    dim3 grid(OD_ / 32, OD_ / 32, 4);       // (16, 16, 4)
    transpose_w_kernel<<<grid, 256, 0, stream>>>(Wq, Wk, Wv, Wo, Wt);
  }
  {
    dim3 grid((B_ * T_ / 32) / 4, H_, 3);   // (64, 8, 3), 4 waves/block
    qkv_kernel<<<grid, 128, 0, stream>>>(Xb, Wt, Q, K, Vt);
  }
  {
    dim3 grid(T_ / 128, B_ * H_);           // (32, 16), 8 waves/block
    attn_kernel<<<grid, 256, 0, stream>>>(Q, K, Vt, attn);
  }
  {
    dim3 grid((B_ * T_ / 32) / 4, OD_ / 64); // (64, 8)
    proj_kernel<<<grid, 128, 0, stream>>>(attn, Wot, bo, out);
  }
}
